// TransformerEncoderLayer_6820408066247
// MI455X (gfx1250) — matrix operand order repaired and checked by reading, not compiled, not run on hardware
//
#include <hip/hip_runtime.h>
#include <hip/hip_bf16.h>
#include <math.h>

// Shapes (compile-time constants from the reference)
#define BB 8
#define SSQ 1024
#define DMm 512
#define HH 8
#define DKk 32
#define DHh 64
#define LN_EPS 1e-5f
#define NEG_SLOPE 0.01f

typedef __bf16 bf16;
typedef bf16  v16bf __attribute__((ext_vector_type(16)));
typedef bf16  v8bf  __attribute__((ext_vector_type(8)));
typedef float v8f   __attribute__((ext_vector_type(8)));

static __device__ inline v8f zero8() {
  v8f z; for (int i = 0; i < 8; ++i) z[i] = 0.f; return z;
}

static __device__ inline v16bf join16(v8bf a, v8bf b) {
  union { v16bf v; v8bf h[2]; } u;
  u.h[0] = a; u.h[1] = b;
  return u.v;
}

// A-fragment (16xK tile, K step 32) from a row-major bf16 matrix.
// Lane L (row = L&15) holds K = k0 + (L>>4)*8 + {0..7} and +16..+23 (ISA 7.12.2).
static __device__ inline v16bf loadA(const bf16* base, int ld, int row0, int k0, int lane) {
  int r = lane & 15, hi = lane >> 4;
  const bf16* p = base + (size_t)(row0 + r) * ld + k0 + hi * 8;
  return join16(*(const v8bf*)p, *(const v8bf*)(p + 16));
}

// B-fragment (32x16 tile) from W stored [N][K] row-major (i.e. X @ W^T GEMMs).
// Lane L (col = L&15) holds K = k0 + (L>>4)*8 + {0..7} and +16..+23 (matches loadA).
static __device__ inline v16bf loadB(const bf16* base, int ld, int n0, int k0, int lane) {
  int n = lane & 15, hi = lane >> 4;
  const bf16* p = base + (size_t)(n0 + n) * ld + k0 + hi * 8;
  return join16(*(const v8bf*)p, *(const v8bf*)(p + 16));
}

static __device__ inline v8f wmma_bf16(v16bf a, v16bf b, v8f c) {
  return __builtin_amdgcn_wmma_f32_16x16x32_bf16(false, a, false, b, (short)0, c, false, false);
}

// ---------------- fp32 -> bf16 conversion ----------------
__global__ void f2b_kernel(const float* __restrict__ src, bf16* __restrict__ dst, int n) {
  int i = blockIdx.x * blockDim.x + threadIdx.x;
  int stride = gridDim.x * blockDim.x;
  for (; i < n; i += stride) dst[i] = (bf16)src[i];
}

// ---------------- Q/K projection + l2norm (2 n-tiles, compile-time) ----------------
// grid: (64 mtile-groups, 8 heads, 2 {q,k}), block 256 = 8 waves, 1 m-tile each.
__global__ void qk_kernel(const bf16* __restrict__ xb,
                          const bf16* __restrict__ Wq_b,
                          const bf16* __restrict__ Wk_b,
                          bf16* __restrict__ qb, bf16* __restrict__ kb) {
  int lane = threadIdx.x & 31, wave = threadIdx.x >> 5;
  int mt = blockIdx.x * 8 + wave;   // 0..511 (B*S/16)
  int h = blockIdx.y;
  int which = blockIdx.z;           // 0=q 1=k
  int m0 = mt * 16;
  const bf16* W = (which == 0 ? Wq_b : Wk_b) + (size_t)h * DKk * DMm;
  v8f c0 = zero8(), c1 = zero8();
#pragma unroll
  for (int k0 = 0; k0 < DMm; k0 += 32) {
    v16bf a = loadA(xb, DMm, m0, k0, lane);
    c0 = wmma_bf16(a, loadB(W, DMm, 0, k0, lane), c0);
    c1 = wmma_bf16(a, loadB(W, DMm, 16, k0, lane), c1);
  }
  int r_ = lane & 15, hi = lane >> 4;
  bf16* out = (which == 0 ? qb : kb) + (size_t)h * (BB * SSQ) * DKk;
#pragma unroll
  for (int r = 0; r < 8; ++r) {
    float ss = c0[r] * c0[r] + c1[r] * c1[r];
    ss += __shfl_xor(ss, 1, 32); ss += __shfl_xor(ss, 2, 32);
    ss += __shfl_xor(ss, 4, 32); ss += __shfl_xor(ss, 8, 32);
    float scale = 1.0f / fmaxf(sqrtf(ss), 1e-12f);
    int m = m0 + r + 8 * hi;
    out[(size_t)m * DKk + r_]      = (bf16)(c0[r] * scale);
    out[(size_t)m * DKk + 16 + r_] = (bf16)(c1[r] * scale);
  }
}

// ---------------- V projection, transpose-store (4 n-tiles, compile-time) ----------------
// grid: (64 mtile-groups, 8 heads), block 256 = 8 waves.
__global__ void v_kernel(const bf16* __restrict__ xb, const bf16* __restrict__ Wv_b,
                         bf16* __restrict__ vbT) {
  int lane = threadIdx.x & 31, wave = threadIdx.x >> 5;
  int mt = blockIdx.x * 8 + wave;
  int h = blockIdx.y;
  int m0 = mt * 16;
  const bf16* W = Wv_b + (size_t)h * DHh * DMm;
  v8f c[4];
#pragma unroll
  for (int i = 0; i < 4; ++i) c[i] = zero8();
#pragma unroll
  for (int k0 = 0; k0 < DMm; k0 += 32) {
    v16bf a = loadA(xb, DMm, m0, k0, lane);
#pragma unroll
    for (int nt = 0; nt < 4; ++nt)
      c[nt] = wmma_bf16(a, loadB(W, DMm, nt * 16, k0, lane), c[nt]);
  }
  int r_ = lane & 15, hi = lane >> 4;
  int bidx = m0 >> 10, s0 = m0 & (SSQ - 1);
  bf16* out = vbT + ((size_t)h * BB + bidx) * DHh * SSQ;  // vbT[h][b][dh][s]
#pragma unroll
  for (int nt = 0; nt < 4; ++nt)
#pragma unroll
    for (int r = 0; r < 8; ++r) {
      int dh = nt * 16 + r_;
      int s = s0 + r + 8 * hi;
      out[(size_t)dh * SSQ + s] = (bf16)c[nt][r];
    }
}

// ---------------- fused causal attention ----------------
// grid: (8 stile-groups, H, B), block 256 = 8 waves; wave owns one 16-row s-tile.
__global__ void attn_kernel(const bf16* __restrict__ qb, const bf16* __restrict__ kb,
                            const bf16* __restrict__ vbT,
                            const float* __restrict__ pd, const float* __restrict__ omega,
                            bf16* __restrict__ yall) {
  __shared__ __align__(32) bf16 lp[8][16][32];  // per-wave 16x32 prob tile
  int lane = threadIdx.x & 31, wave = threadIdx.x >> 5;
  int stile = blockIdx.x * 8 + wave;  // 0..63
  int h = blockIdx.y, b = blockIdx.z;
  int s0 = stile * 16;
  int r_ = lane & 15, hi = lane >> 4;
  float om = omega[h];
  const bf16* qh = qb + ((size_t)h * BB + b) * SSQ * DKk;
  const bf16* kh = kb + ((size_t)h * BB + b) * SSQ * DKk;
  const bf16* vh = vbT + ((size_t)h * BB + b) * DHh * SSQ;
  const float* pdb = pd + (size_t)b * SSQ * SSQ;

  v16bf aQ = loadA(qh, DKk, s0, 0, lane);  // q rows, full K=32
  float zz[8];
#pragma unroll
  for (int i = 0; i < 8; ++i) zz[i] = 0.f;
  v8f acc0 = zero8(), acc1 = zero8(), acc2 = zero8(), acc3 = zero8();

  for (int t0 = 0; t0 <= s0 + 15; t0 += 32) {
    // scores for two 16-wide t-tiles -> exp(score + exp(-omega*d)-1) -> LDS (bf16)
#pragma unroll
    for (int sub = 0; sub < 2; ++sub) {
      int tb = t0 + sub * 16;
      if (tb <= s0 + 15) {
        v16bf bK = loadB(kh, DKk, tb, 0, lane);
        v8f sc = wmma_bf16(aQ, bK, zero8());  // K=DK=32: single WMMA
#pragma unroll
        for (int r = 0; r < 8; ++r) {
          int s = s0 + r + 8 * hi;
          int t = tb + r_;
          float e = 0.f;
          if (t <= s) {  // causal mask (reference mask is triu(k=1))
            float d = pdb[(size_t)s * SSQ + t];
            e = __expf(sc[r] + __expf(-om * d) - 1.0f);
          }
          zz[r] += e;
          lp[wave][r + 8 * hi][sub * 16 + r_] = (bf16)e;
        }
      } else {
#pragma unroll
        for (int r = 0; r < 8; ++r) lp[wave][r + 8 * hi][sub * 16 + r_] = (bf16)0.f;
      }
    }
    // re-shape probs C-fragment -> A-fragment via LDS, then 4 PV WMMAs
    v16bf aP = loadA(&lp[wave][0][0], 32, 0, 0, lane);
    acc0 = wmma_bf16(aP, loadB(vh, SSQ, 0,  t0, lane), acc0);
    acc1 = wmma_bf16(aP, loadB(vh, SSQ, 16, t0, lane), acc1);
    acc2 = wmma_bf16(aP, loadB(vh, SSQ, 32, t0, lane), acc2);
    acc3 = wmma_bf16(aP, loadB(vh, SSQ, 48, t0, lane), acc3);
  }

  // custom softmax denominator: sum + 1
#pragma unroll
  for (int r = 0; r < 8; ++r) {
    float z = zz[r];
    z += __shfl_xor(z, 1, 32); z += __shfl_xor(z, 2, 32);
    z += __shfl_xor(z, 4, 32); z += __shfl_xor(z, 8, 32);
    zz[r] = 1.0f / (z + 1.0f);
  }
  // heads concatenated: yall[b][s][h*64+dh] (bf16, feeds Wo GEMM as A)
  bf16* yb = yall + ((size_t)b * SSQ + s0) * DMm + h * DHh;
#pragma unroll
  for (int r = 0; r < 8; ++r) {
    int s = r + 8 * hi;
    yb[(size_t)s * DMm + r_]      = (bf16)(acc0[r] * zz[r]);
    yb[(size_t)s * DMm + 16 + r_] = (bf16)(acc1[r] * zz[r]);
    yb[(size_t)s * DMm + 32 + r_] = (bf16)(acc2[r] * zz[r]);
    yb[(size_t)s * DMm + 48 + r_] = (bf16)(acc3[r] * zz[r]);
  }
}

// ---------------- output projection + residual (pre-LN1) ----------------
// Each wave: 16x64 strip (4 accumulators, A reused 4x).
// waves = 512 m-tiles * 8 n-groups = 4096 -> grid 512 x 256
__global__ void proj_kernel(const bf16* __restrict__ yall, const bf16* __restrict__ Wo_b,
                            const float* __restrict__ bo, const float* __restrict__ x,
                            float* __restrict__ z1pre) {
  int lane = threadIdx.x & 31, wave = threadIdx.x >> 5;
  int w = blockIdx.x * 8 + wave;
  int mt = w >> 3, ng = w & 7;
  int m0 = mt * 16, n0 = ng * 64;
  v8f c[4];
#pragma unroll
  for (int i = 0; i < 4; ++i) c[i] = zero8();
#pragma unroll
  for (int k0 = 0; k0 < DMm; k0 += 32) {
    v16bf a = loadA(yall, DMm, m0, k0, lane);
#pragma unroll
    for (int nt = 0; nt < 4; ++nt)
      c[nt] = wmma_bf16(a, loadB(Wo_b, DMm, n0 + nt * 16, k0, lane), c[nt]);
  }
  int r_ = lane & 15, hi = lane >> 4;
#pragma unroll
  for (int nt = 0; nt < 4; ++nt) {
    int n = n0 + nt * 16 + r_;
    float bias = bo[n];
#pragma unroll
    for (int r = 0; r < 8; ++r) {
      int m = m0 + r + 8 * hi;
      z1pre[(size_t)m * DMm + n] = c[nt][r] + bias + x[(size_t)m * DMm + n];
    }
  }
}

// ---------------- FFN GEMM + LeakyReLU + residual (pre-LN2) ----------------
__global__ void ffn_kernel(const bf16* __restrict__ z1b, const bf16* __restrict__ Wf_b,
                           const float* __restrict__ bf_, const float* __restrict__ z1,
                           float* __restrict__ hpre) {
  int lane = threadIdx.x & 31, wave = threadIdx.x >> 5;
  int w = blockIdx.x * 8 + wave;
  int mt = w >> 3, ng = w & 7;
  int m0 = mt * 16, n0 = ng * 64;
  v8f c[4];
#pragma unroll
  for (int i = 0; i < 4; ++i) c[i] = zero8();
#pragma unroll
  for (int k0 = 0; k0 < DMm; k0 += 32) {
    v16bf a = loadA(z1b, DMm, m0, k0, lane);
#pragma unroll
    for (int nt = 0; nt < 4; ++nt)
      c[nt] = wmma_bf16(a, loadB(Wf_b, DMm, n0 + nt * 16, k0, lane), c[nt]);
  }
  int r_ = lane & 15, hi = lane >> 4;
#pragma unroll
  for (int nt = 0; nt < 4; ++nt) {
    int n = n0 + nt * 16 + r_;
    float bias = bf_[n];
#pragma unroll
    for (int r = 0; r < 8; ++r) {
      int m = m0 + r + 8 * hi;
      float t = c[nt][r] + bias;
      t = t > 0.f ? t : NEG_SLOPE * t;
      hpre[(size_t)m * DMm + n] = z1[(size_t)m * DMm + n] + t;
    }
  }
}

// ---------------- LayerNorm over DM=512 (one wave per row) ----------------
__global__ void ln_kernel(const float* __restrict__ src, const float* __restrict__ w,
                          const float* __restrict__ bias, float* __restrict__ out_f,
                          bf16* __restrict__ out_b) {
  int lane = threadIdx.x & 31, wave = threadIdx.x >> 5;
  int row = blockIdx.x * 8 + wave;
  const float* p = src + (size_t)row * DMm;
  float v[16];
  float sum = 0.f;
#pragma unroll
  for (int j = 0; j < 16; ++j) { v[j] = p[lane + j * 32]; sum += v[j]; }
#pragma unroll
  for (int m = 1; m < 32; m <<= 1) sum += __shfl_xor(sum, m, 32);
  float mu = sum * (1.0f / DMm);
  float var = 0.f;
#pragma unroll
  for (int j = 0; j < 16; ++j) { float d = v[j] - mu; var += d * d; }
#pragma unroll
  for (int m = 1; m < 32; m <<= 1) var += __shfl_xor(var, m, 32);
  var *= (1.0f / DMm);
  float rs = rsqrtf(var + LN_EPS);
#pragma unroll
  for (int j = 0; j < 16; ++j) {
    int i = lane + j * 32;
    float o = (v[j] - mu) * rs * w[i] + bias[i];
    out_f[(size_t)row * DMm + i] = o;
    if (out_b) out_b[(size_t)row * DMm + i] = (bf16)o;
  }
}

extern "C" void kernel_launch(void* const* d_in, const int* in_sizes, int n_in,
                              void* d_out, int out_size, void* d_ws, size_t ws_size,
                              hipStream_t stream) {
  const float* x     = (const float*)d_in[0];
  // d_in[1] = attn_mask (bool, causal triu) -> applied structurally
  const float* pd    = (const float*)d_in[2];
  const float* Wq    = (const float*)d_in[3];
  const float* Wk    = (const float*)d_in[4];
  const float* Wv    = (const float*)d_in[5];
  const float* omega = (const float*)d_in[6];
  const float* Wo    = (const float*)d_in[7];
  const float* bo    = (const float*)d_in[8];
  const float* Wf    = (const float*)d_in[9];
  const float* bf_   = (const float*)d_in[10];
  const float* ln1w  = (const float*)d_in[11];
  const float* ln1b  = (const float*)d_in[12];
  const float* ln2w  = (const float*)d_in[13];
  const float* ln2b  = (const float*)d_in[14];
  float* out = (float*)d_out;

  const int MROWS = BB * SSQ;  // 8192
  char* ws = (char*)d_ws;
  size_t off = 0;
  auto alloc = [&](size_t bytes) { char* p = ws + off; off += (bytes + 255) & ~(size_t)255; return p; };

  bf16* xb   = (bf16*)alloc((size_t)MROWS * DMm * 2);            // 8 MB
  bf16* Wq_b = (bf16*)alloc((size_t)HH * DKk * DMm * 2);         // .25 MB
  bf16* Wk_b = (bf16*)alloc((size_t)HH * DKk * DMm * 2);
  bf16* Wv_b = (bf16*)alloc((size_t)HH * DHh * DMm * 2);         // .5 MB
  bf16* Wo_b = (bf16*)alloc((size_t)DMm * DMm * 2);              // .5 MB
  bf16* Wf_b = (bf16*)alloc((size_t)DMm * DMm * 2);
  bf16* qb   = (bf16*)alloc((size_t)HH * MROWS * DKk * 2);       // 4 MB
  bf16* kb   = (bf16*)alloc((size_t)HH * MROWS * DKk * 2);
  bf16* vbT  = (bf16*)alloc((size_t)HH * BB * DHh * SSQ * 2);    // 8 MB
  bf16* yall = (bf16*)alloc((size_t)MROWS * DMm * 2);            // 8 MB
  float* z1pre = (float*)alloc((size_t)MROWS * DMm * 4);         // 16 MB (reused as hpre)
  float* z1    = (float*)alloc((size_t)MROWS * DMm * 4);         // 16 MB
  bf16* z1b  = (bf16*)alloc((size_t)MROWS * DMm * 2);            // 8 MB
  float* hpre = z1pre;  // LN1 consumed z1pre; alias for FFN output

  // 1) fp32 -> bf16 staging (weights stay [N][K] row-major = ready B matrices)
  f2b_kernel<<<4096, 256, 0, stream>>>(x,  xb,   MROWS * DMm);
  f2b_kernel<<<512,  256, 0, stream>>>(Wq, Wq_b, HH * DKk * DMm);
  f2b_kernel<<<512,  256, 0, stream>>>(Wk, Wk_b, HH * DKk * DMm);
  f2b_kernel<<<1024, 256, 0, stream>>>(Wv, Wv_b, HH * DHh * DMm);
  f2b_kernel<<<1024, 256, 0, stream>>>(Wo, Wo_b, DMm * DMm);
  f2b_kernel<<<1024, 256, 0, stream>>>(Wf, Wf_b, DMm * DMm);

  // 2) Q/K projection (+l2norm) and V projection (+transpose)
  qk_kernel<<<dim3(64, HH, 2), 256, 0, stream>>>(xb, Wq_b, Wk_b, qb, kb);
  v_kernel<<<dim3(64, HH), 256, 0, stream>>>(xb, Wv_b, vbT);

  // 3) fused causal attention (scores + bias + exp + PV, single pass)
  attn_kernel<<<dim3(8, HH, BB), 256, 0, stream>>>(qb, kb, vbT, pd, omega, yall);

  // 4) output projection + bias + residual
  proj_kernel<<<512, 256, 0, stream>>>(yall, Wo_b, bo, x, z1pre);

  // 5) LN1 -> z1 (fp32) + z1b (bf16 for FFN GEMM)
  ln_kernel<<<1024, 256, 0, stream>>>(z1pre, ln1w, ln1b, z1, z1b);

  // 6) FFN GEMM + LeakyReLU + residual
  ffn_kernel<<<512, 256, 0, stream>>>(z1b, Wf_b, bf_, z1, hpre);

  // 7) LN2 -> final output
  ln_kernel<<<1024, 256, 0, stream>>>(hpre, ln2w, ln2b, out, nullptr);

  (void)in_sizes; (void)n_in; (void)out_size; (void)ws_size;
}
